// Bidirectional_LSTM_2869038154476
// MI455X (gfx1250) — matrix operand order repaired and checked by reading, not compiled, not run on hardware
//
#include <hip/hip_runtime.h>
#include <hip/hip_bf16.h>

// ---------------------------------------------------------------------------
// Bidirectional LSTM for MI455X (gfx1250), persistent-register recurrence.
//   B=32, T=2048, D=256, U=256.  out: [B, T, 2U] fp32.
//
// grid = 4 persistent WGs (dir x batch-half), 512 threads = 16 waves.
// Wave w owns permuted N-tiles {4w..4w+3} == hidden units [16w, 16w+16),
// all four gates (gate-interleaved column permutation baked into the pack).
// Uh fragments for 2 of the 4 tiles live in VGPRs for all 2048 steps; the
// rest (and all W fragments) stream from L2 each step. The streamed base
// pointers are laundered through an empty asm per iteration (blocks LICM
// hoist+spill) as explicit address_space(1) pointers (keeps GLOBAL_LOAD
// lowering with separate LOADcnt, instead of FLAT which also ticks DScnt
// and forces combined loadcnt+dscnt drains before every WMMA).
// ---------------------------------------------------------------------------

typedef __attribute__((ext_vector_type(16))) __bf16 v16bf;
typedef __attribute__((ext_vector_type(8)))  __bf16 bf16x8;
typedef __attribute__((ext_vector_type(8)))  float  v8f;

#define GAS __attribute__((address_space(1)))

union FragA { v16bf v; bf16x8 h[2]; };

#define LSTM_B   32
#define LSTM_T   2048
#define LSTM_D   256
#define LSTM_U   256
#define NGATE    1024          // 4*U, permuted: p -> (gate = p&3, unit = p>>2)
#define KT       8             // K tiles of 32 (K = 256)
#define NT       64            // N tiles of 16 (N = 1024)
#define FRAG_ELEMS (KT*NT*32*16)   // 262144 bf16 per packed matrix
#define XSTR     264           // padded LDS row stride (bf16 elems), 16B multiple
#define NTW      4             // N-tiles per wave
#define NRES     2             // of which VGPR-resident (Uh)

__device__ __forceinline__ unsigned short f32_to_bf16_bits(float f) {
    unsigned int u = __builtin_bit_cast(unsigned int, f);
    u = (u + 0x7fffu + ((u >> 16) & 1u)) >> 16;     // round-to-nearest-even
    return (unsigned short)u;
}
__device__ __forceinline__ float sigm(float x)  { return 1.0f / (1.0f + __expf(-x)); }
__device__ __forceinline__ float tanh_(float x) { float e = __expf(-2.0f * x); return (1.0f - e) / (1.0f + e); }

// ---------------------------------------------------------------------------
// Pack W_f/W_b/U_f/U_b ([256][1024] f32, row-major) into bf16 WMMA B-fragments,
// lane-major, gate-interleaved column permutation.
//   B-frag (16x16x32 bf16): lane&15 = n_local, K = (half&7) + 8*(lane>>4) + 16*(half>>3).
//   dst element e = ((kt*64 + ntile)*32 + lane)*16 + half.
//   sel: 0=W_f 1=W_b 2=U_f 3=U_b   at packed + sel*FRAG_ELEMS.
// ---------------------------------------------------------------------------
__global__ __launch_bounds__(256) void pack_weights(
    const float* __restrict__ Wf, const float* __restrict__ Uf,
    const float* __restrict__ Wb, const float* __restrict__ Ub,
    unsigned short* __restrict__ packed)
{
    int idx = blockIdx.x * 256 + threadIdx.x;        // 0 .. 4*FRAG_ELEMS-1
    int e   = idx & (FRAG_ELEMS - 1);
    int sel = idx >> 18;
    const float* src = (sel == 0) ? Wf : (sel == 1) ? Wb : (sel == 2) ? Uf : Ub;

    int half  = e & 15;
    int lane  = (e >> 4)  & 31;
    int ntile = (e >> 9)  & 63;
    int kt    = e >> 15;

    int K     = kt * 32 + (half & 7) + ((lane >> 4) << 3) + ((half >> 3) << 4);
    int p     = ntile * 16 + (lane & 15);            // permuted column
    int gate  = p & 3;
    int n     = p >> 2;
    float v   = src[K * NGATE + (gate << 8) + n];
    packed[idx] = f32_to_bf16_bits(v);
}

// ---------------------------------------------------------------------------
__global__ __launch_bounds__(512, 1) void bilstm_scan(
    const float* __restrict__ x,           // [B][T][D] f32
    const float* __restrict__ bias_f,      // [4U]
    const float* __restrict__ bias_b,      // [4U]
    const unsigned short* __restrict__ packed,
    float* __restrict__ out)               // [B][T][2U]
{
    __shared__ __align__(16) unsigned short xbuf[16 * XSTR];   // x_t rows, bf16
    __shared__ __align__(16) unsigned short hbuf[16 * XSTR];   // h_{t-1},  bf16

    const int dir   = (int)blockIdx.x >> 1;
    const int b0    = ((int)blockIdx.x & 1) * 16;
    const int tid   = (int)threadIdx.x;
    const int wave  = tid >> 5;
    const int lane  = tid & 31;
    const int nt0   = wave * NTW;

    // explicit global-address-space views of the packed weight buffers
    const GAS unsigned short* pW =
        (const GAS unsigned short*)(packed + (size_t)dir * FRAG_ELEMS);       // sel 0/1
    const GAS unsigned short* pU =
        (const GAS unsigned short*)(packed + (size_t)(2 + dir) * FRAG_ELEMS); // sel 2/3

    // per-(wave,lane) element offsets of streamed fragments (loop-invariant)
    int offW[KT][NTW];
    int offU[KT][NTW - NRES];
    #pragma unroll
    for (int kt = 0; kt < KT; ++kt) {
        #pragma unroll
        for (int tt = 0; tt < NTW; ++tt)
            offW[kt][tt] = (((kt * NT) + nt0 + tt) * 32 + lane) * 16;
        #pragma unroll
        for (int tt = NRES; tt < NTW; ++tt)
            offU[kt][tt - NRES] = (((kt * NT) + nt0 + tt) * 32 + lane) * 16;
    }

    // ---- persistent Uh B-fragments for tiles nt0, nt0+1 (128 VGPRs/lane) ----
    FragA ub[KT][NRES];
    #pragma unroll
    for (int kt = 0; kt < KT; ++kt)
        #pragma unroll
        for (int tt = 0; tt < NRES; ++tt) {
            const GAS bf16x8* p =
                (const GAS bf16x8*)(pU + (((kt * NT) + nt0 + tt) * 32 + lane) * 16);
            ub[kt][tt].h[0] = p[0];
            ub[kt][tt].h[1] = p[1];
        }

    // ---- bias per lane per tile (permuted gate layout) ----
    const float* bias = dir ? bias_b : bias_f;
    float bz[NTW];
    #pragma unroll
    for (int tt = 0; tt < NTW; ++tt) {
        int p  = (nt0 + tt) * 16 + (lane & 15);
        bz[tt] = bias[((p & 3) << 8) + (p >> 2)];
    }

    // ---- init: h = 0 in LDS, c = 0 in registers ----
    for (int i = tid; i < 16 * XSTR; i += 512) hbuf[i] = 0;
    float cst[NTW][8];
    #pragma unroll
    for (int tt = 0; tt < NTW; ++tt)
        #pragma unroll
        for (int v = 0; v < 8; ++v) cst[tt][v] = 0.0f;
    __syncthreads();

    const int  row_a    = lane & 15;                 // A-fragment row
    const int  ksel     = (lane >> 4) ? 8 : 0;       // A-fragment K sub-offset
    const bool use_tanh = (dir == 0);
    const int  lg       = lane & ~3;                 // 4-lane gate group leader

    for (int t = 0; t < LSTM_T; ++t) {
        const int tx = dir ? (LSTM_T - 1 - t) : t;   // time index into x / out

        // Launder streamed-fragment base pointers (still address_space(1)):
        // opaque per iteration => loads stay in-loop as global_load_b128.
        const GAS unsigned short* pWt = pW;
        const GAS unsigned short* pUt = pU;
        asm volatile("" : "+s"(pWt), "+s"(pUt));

        // ---- stage x[b0..b0+15, tx, :] into LDS as bf16 ----
        {
            const float* xsrc = x + ((size_t)b0 * LSTM_T + tx) * LSTM_D;
            #pragma unroll
            for (int j = 0; j < 8; ++j) {
                int e = tid + j * 512;
                int r = e >> 8, c = e & 255;
                float xv = xsrc[(size_t)r * LSTM_T * LSTM_D + c];
                xbuf[r * XSTR + c] = f32_to_bf16_bits(xv);
            }
            if (t + 1 < LSTM_T) {                    // prefetch next step's rows
                const int txn = dir ? (LSTM_T - 2 - t) : t + 1;
                const float* xn = x + ((size_t)b0 * LSTM_T + txn) * LSTM_D;
                int r = tid >> 5, c = (tid & 31) * 8;
                __builtin_prefetch(xn + (size_t)r * LSTM_T * LSTM_D + c, 0, 1);
            }
        }
        __syncthreads();   // publishes xbuf AND previous step's hbuf writes

        v8f acc[NTW] = {};

        // ---- z += x_t @ W   (all W B-fragments streamed from L2) ----
        #pragma unroll
        for (int kt = 0; kt < KT; ++kt) {
            FragA a;
            const unsigned short* ap = &xbuf[row_a * XSTR + kt * 32 + ksel];
            a.h[0] = *(const bf16x8*)(ap);
            a.h[1] = *(const bf16x8*)(ap + 16);
            #pragma unroll
            for (int tt = 0; tt < NTW; ++tt) {
                FragA w;
                const GAS bf16x8* p = (const GAS bf16x8*)(pWt + offW[kt][tt]);
                w.h[0] = p[0];
                w.h[1] = p[1];
                acc[tt] = __builtin_amdgcn_wmma_f32_16x16x32_bf16(
                    false, a.v, false, w.v, (short)0, acc[tt], false, false);
            }
        }

        // ---- z += h_{t-1} @ Uh  (tiles 0,1 VGPR-resident; 2,3 L2 stream) ----
        #pragma unroll
        for (int kt = 0; kt < KT; ++kt) {
            FragA a;
            const unsigned short* ap = &hbuf[row_a * XSTR + kt * 32 + ksel];
            a.h[0] = *(const bf16x8*)(ap);
            a.h[1] = *(const bf16x8*)(ap + 16);
            #pragma unroll
            for (int tt = 0; tt < NRES; ++tt)
                acc[tt] = __builtin_amdgcn_wmma_f32_16x16x32_bf16(
                    false, a.v, false, ub[kt][tt].v, (short)0, acc[tt], false, false);
            #pragma unroll
            for (int tt = NRES; tt < NTW; ++tt) {
                FragA u;
                const GAS bf16x8* p = (const GAS bf16x8*)(pUt + offU[kt][tt - NRES]);
                u.h[0] = p[0];
                u.h[1] = p[1];
                acc[tt] = __builtin_amdgcn_wmma_f32_16x16x32_bf16(
                    false, a.v, false, u.v, (short)0, acc[tt], false, false);
            }
        }

        __syncthreads();   // all hbuf reads done before epilogue rewrites it

        // ---- gate epilogue: 4-lane shuffle gathers i,f,g,o per unit ----
        #pragma unroll
        for (int tt = 0; tt < NTW; ++tt) {
            const int pcol = (nt0 + tt) * 16 + (lane & 15);
            const int n    = pcol >> 2;
            #pragma unroll
            for (int v = 0; v < 8; ++v) {
                float z  = acc[tt][v] + bz[tt];
                float zi = __shfl(z, lg + 0, 32);
                float zf = __shfl(z, lg + 1, 32);
                float zg = __shfl(z, lg + 2, 32);
                float zo = __shfl(z, lg + 3, 32);
                float gi = sigm(zi), gf = sigm(zf), go = sigm(zo);
                float gg = use_tanh ? tanh_(zg) : fmaxf(zg, 0.0f);
                float c  = gf * cst[tt][v] + gi * gg;
                cst[tt][v] = c;
                float ca = use_tanh ? tanh_(c) : fmaxf(c, 0.0f);
                float h  = go * ca;
                if ((lane & 3) == 0) {
                    int row = v + ((lane >> 4) << 3);
                    hbuf[row * XSTR + n] = f32_to_bf16_bits(h);
                    out[((size_t)(b0 + row) * LSTM_T + tx) * (2 * LSTM_U)
                        + dir * LSTM_U + n] = h;
                }
            }
        }
        // next iteration's __syncthreads (after xbuf fill) orders hbuf writes
        // before the next GEMM's reads.
    }
}

// ---------------------------------------------------------------------------
extern "C" void kernel_launch(void* const* d_in, const int* in_sizes, int n_in,
                              void* d_out, int out_size, void* d_ws, size_t ws_size,
                              hipStream_t stream) {
    const float* x  = (const float*)d_in[0];
    const float* Wf = (const float*)d_in[1];
    const float* Uf = (const float*)d_in[2];
    const float* bf = (const float*)d_in[3];
    const float* Wb = (const float*)d_in[4];
    const float* Ub = (const float*)d_in[5];
    const float* bb = (const float*)d_in[6];
    float* out = (float*)d_out;
    unsigned short* packed = (unsigned short*)d_ws;   // 4 * 512KB = 2MB

    pack_weights<<<(4 * FRAG_ELEMS) / 256, 256, 0, stream>>>(Wf, Uf, Wb, Ub, packed);
    bilstm_scan<<<4, 512, 0, stream>>>(x, bf, bb, packed, out);
}
